// YourNewModelAttention_7662221656632
// MI455X (gfx1250) — matrix operand order repaired and checked by reading, not compiled, not run on hardware
//
#include <hip/hip_runtime.h>
#include <hip/hip_bf16.h>

typedef _Float16 h16;
typedef __attribute__((ext_vector_type(16))) _Float16 v16h;
typedef __attribute__((ext_vector_type(8)))  float    v8f;

#define Bb  2
#define Ss  2048
#define Dd  1024
#define Hh  16
#define HDd 64
#define MM  (Bb * Ss)          // 4096 rows

// ---------------------------------------------------------------------------
// WMMA helpers (CDNA5: D = A(16x32,f16) * B(32x16,f16) + C(16x16,f32))
// ---------------------------------------------------------------------------
__device__ __forceinline__ v8f wmma_f16(v16h a, v16h b, v8f c) {
  return __builtin_amdgcn_wmma_f32_16x16x32_f16(
      /*neg_a=*/false, a, /*neg_b=*/false, b,
      /*c_mod=*/(short)0, c, /*reuse_a=*/false, /*reuse_b=*/false);
}

// A fragment (16x32, f16, row-major source, ld = lda):
//   lane&15 = row M; lane>>4 selects K-halves {0..7,16..23} vs {8..15,24..31}
__device__ __forceinline__ v16h load_a16(const h16* A, int lda, int m0, int k0,
                                         int lane) {
  int m  = m0 + (lane & 15);
  int kh = (lane >> 4) * 8;
  const h16* p0 = A + (size_t)m * lda + k0 + kh;
  v16h f;
#pragma unroll
  for (int j = 0; j < 8; ++j) { f[j] = p0[j]; f[8 + j] = p0[16 + j]; }
  return f;
}

// B fragment (32x16, f16) from a [N][K] (transposed) source, ld = ldb:
//   lane&15 = column N; lane>>4 selects K-halves {0..7,16..23} vs {8..15,24..31}
__device__ __forceinline__ v16h load_b16(const h16* Bt, int ldb, int n0, int k0,
                                         int lane) {
  int n  = n0 + (lane & 15);
  int kk = k0 + (lane >> 4) * 8;
  const h16* p = Bt + (size_t)n * ldb + kk;
  v16h f;
#pragma unroll
  for (int j = 0; j < 8; ++j) { f[j] = p[j]; f[8 + j] = p[16 + j]; }
  return f;
}

// ---------------------------------------------------------------------------
// fp32 -> f16 convert, and convert+transpose for weights
// ---------------------------------------------------------------------------
__global__ __launch_bounds__(256) void cvt_f16(const float* __restrict__ x,
                                               h16* __restrict__ y, int n) {
  int i = blockIdx.x * blockDim.x + threadIdx.x;
  if (i < n) y[i] = (h16)x[i];
}

// src [rows][cols] fp32  ->  dst [cols][rows] f16
__global__ __launch_bounds__(256) void cvt_transpose(
    const float* __restrict__ src, h16* __restrict__ dst, int rows, int cols) {
  int i = blockIdx.x * blockDim.x + threadIdx.x;
  if (i < rows * cols) {
    int r = i / cols, c = i % cols;
    dst[(size_t)c * rows + r] = (h16)src[i];
  }
}

// ---------------------------------------------------------------------------
// GEMM 1: qkv = hidden(4096x1024) @ Wqkv(1024x3072) + b
//   Register-blocked: each wave computes a 32x64 tile (2 M-tiles x 4 N-tiles),
//   8 WMMAs per 32-wide K-chunk reusing 2 A-frags and 4 B-frags.
//   scatter store: Q,K -> [b][h][s][d] ; V -> [b][h][d][s] (transposed)
// ---------------------------------------------------------------------------
__global__ __launch_bounds__(256) void gemm_qkv(
    const h16* __restrict__ A, const h16* __restrict__ Bt,
    const float* __restrict__ bias, h16* __restrict__ q, h16* __restrict__ k,
    h16* __restrict__ vt) {
  const int NTg = (3 * Dd) / 64;                      // 48 column super-tiles
  int wave = threadIdx.x >> 5, lane = threadIdx.x & 31;
  int id = blockIdx.x * 8 + wave;
  int m0 = (id / NTg) * 32, n0 = (id % NTg) * 64;

  v8f acc[2][4];
#pragma unroll
  for (int mi = 0; mi < 2; ++mi)
#pragma unroll
    for (int j = 0; j < 4; ++j) acc[mi][j] = v8f{};

  for (int kk = 0; kk < Dd; kk += 32) {
    __builtin_prefetch(A + (size_t)(m0 + (lane & 15)) * Dd + kk + 64, 0, 0);
    __builtin_prefetch(Bt + (size_t)(n0 + (lane & 15)) * Dd + kk + 64, 0, 0);
    v16h a0 = load_a16(A, Dd, m0, kk, lane);
    v16h a1 = load_a16(A, Dd, m0 + 16, kk, lane);
#pragma unroll
    for (int j = 0; j < 4; ++j) {
      v16h b = load_b16(Bt, Dd, n0 + 16 * j, kk, lane);
      acc[0][j] = wmma_f16(a0, b, acc[0][j]);
      acc[1][j] = wmma_f16(a1, b, acc[1][j]);
    }
  }

  int lg = lane & 15, hi = lane >> 4;
#pragma unroll
  for (int j = 0; j < 4; ++j) {
    int n = n0 + 16 * j + lg;
    float bv = bias[n];
#pragma unroll
    for (int mi = 0; mi < 2; ++mi) {
#pragma unroll
      for (int i = 0; i < 8; ++i) {
        int r = m0 + mi * 16 + (hi << 3) + i;
        float val = acc[mi][j][i] + bv;
        int bb = r >> 11, s = r & (Ss - 1);
        if (n < Dd) {
          int h = n >> 6, dd = n & 63;
          q[(((size_t)(bb * Hh + h) * Ss + s) << 6) + dd] = (h16)val;
        } else if (n < 2 * Dd) {
          int nn = n - Dd, h = nn >> 6, dd = nn & 63;
          k[(((size_t)(bb * Hh + h) * Ss + s) << 6) + dd] = (h16)val;
        } else {
          int nn = n - 2 * Dd, h = nn >> 6, dd = nn & 63;
          vt[((size_t)(bb * Hh + h) * HDd + dd) * Ss + s] = (h16)val;
        }
      }
    }
  }
}

// ---------------------------------------------------------------------------
// Flash attention: one wave owns a 32-row Q tile-pair of one (b,h).
//   Per 32-key chunk: 8 score WMMAs (K-frags reused across both M-tiles),
//   online softmax with shfl_xor 16-lane reductions, P via LDS (f32 C-layout
//   -> f16 A-layout, explicit s_wait_dscnt), then 8 PV WMMAs (V-frags reused).
// ---------------------------------------------------------------------------
__global__ __launch_bounds__(256) void attn_flash(
    const h16* __restrict__ Q, const h16* __restrict__ K,
    const h16* __restrict__ VT, h16* __restrict__ O) {
  __shared__ h16 lds[8][32 * 32];      // per-wave 32 rows x 32 keys of P
  int wave = threadIdx.x >> 5, lane = threadIdx.x & 31;
  int id = blockIdx.x * 8 + wave;
  int qt = id & 63;          // S/32 = 64 q tile-pairs per head
  int bh = id >> 6;          // b*H + h
  const h16* Qp = Q + (size_t)bh * Ss * HDd;
  const h16* Kp = K + (size_t)bh * Ss * HDd;
  const h16* Vp = VT + (size_t)bh * HDd * Ss;
  int q0 = qt * 32;
  int lg = lane & 15, hi = lane >> 4;

  v16h aq[2][2];             // [mtile][k-chunk of HD]
#pragma unroll
  for (int mi = 0; mi < 2; ++mi) {
    aq[mi][0] = load_a16(Qp, HDd, q0 + mi * 16, 0, lane);
    aq[mi][1] = load_a16(Qp, HDd, q0 + mi * 16, 32, lane);
  }

  float mrow[2][8], lrow[2][8];
  v8f acc[2][4];
#pragma unroll
  for (int mi = 0; mi < 2; ++mi) {
#pragma unroll
    for (int t = 0; t < 4; ++t) acc[mi][t] = v8f{};
#pragma unroll
    for (int i = 0; i < 8; ++i) { mrow[mi][i] = -1e30f; lrow[mi][i] = 0.f; }
  }

  const float scl = 0.125f;  // 1/sqrt(64)
  int kend = q0 + 32;
  for (int kc = 0; kc < kend; kc += 32) {
    // ---- scores: S[mi][u] = Q_mi (16x64) @ K_u^T (64x16) --------------
    v16h b00 = load_b16(Kp, HDd, kc, 0, lane);       // keys kc..kc+15
    v16h b01 = load_b16(Kp, HDd, kc, 32, lane);
    v16h b10 = load_b16(Kp, HDd, kc + 16, 0, lane);  // keys kc+16..kc+31
    v16h b11 = load_b16(Kp, HDd, kc + 16, 32, lane);
    v8f s[2][2];
#pragma unroll
    for (int mi = 0; mi < 2; ++mi) {
      s[mi][0] = v8f{};
      s[mi][0] = wmma_f16(aq[mi][0], b00, s[mi][0]);
      s[mi][0] = wmma_f16(aq[mi][1], b01, s[mi][0]);
      s[mi][1] = v8f{};
      s[mi][1] = wmma_f16(aq[mi][0], b10, s[mi][1]);
      s[mi][1] = wmma_f16(aq[mi][1], b11, s[mi][1]);
    }
    // ---- online softmax + stage P into LDS ----------------------------
#pragma unroll
    for (int mi = 0; mi < 2; ++mi) {
#pragma unroll
      for (int i = 0; i < 8; ++i) {
        int row = q0 + mi * 16 + (hi << 3) + i;   // global query index
        int c0 = kc + lg, c1 = kc + 16 + lg;      // key columns of this lane
        float x0 = s[mi][0][i] * scl, x1 = s[mi][1][i] * scl;
        if (c0 > row) x0 = -1e30f;                // causal mask
        if (c1 > row) x1 = -1e30f;
        float rmax = fmaxf(x0, x1);
        rmax = fmaxf(rmax, __shfl_xor(rmax, 1, 32));
        rmax = fmaxf(rmax, __shfl_xor(rmax, 2, 32));
        rmax = fmaxf(rmax, __shfl_xor(rmax, 4, 32));
        rmax = fmaxf(rmax, __shfl_xor(rmax, 8, 32));
        float mnew  = fmaxf(mrow[mi][i], rmax);
        float alpha = __expf(mrow[mi][i] - mnew);
        float p0 = __expf(x0 - mnew), p1 = __expf(x1 - mnew);
        float rs = p0 + p1;
        rs += __shfl_xor(rs, 1, 32);
        rs += __shfl_xor(rs, 2, 32);
        rs += __shfl_xor(rs, 4, 32);
        rs += __shfl_xor(rs, 8, 32);
        lrow[mi][i] = lrow[mi][i] * alpha + rs;
        mrow[mi][i] = mnew;
#pragma unroll
        for (int t = 0; t < 4; ++t) acc[mi][t][i] *= alpha;
        int m = mi * 16 + (hi << 3) + i;          // P row within 32x32 tile
        lds[wave][m * 32 + lg]      = (h16)p0;
        lds[wave][m * 32 + 16 + lg] = (h16)p1;
      }
    }
    asm volatile("s_wait_dscnt 0" ::: "memory");
    // ---- P @ V: V-frags reused across both M-tiles ---------------------
    v16h ap[2];
#pragma unroll
    for (int mi = 0; mi < 2; ++mi) {
      const h16* p0 = &lds[wave][(mi * 16 + lg) * 32 + hi * 8];
#pragma unroll
      for (int j = 0; j < 8; ++j) { ap[mi][j] = p0[j]; ap[mi][8 + j] = p0[16 + j]; }
    }
#pragma unroll
    for (int t = 0; t < 4; ++t) {
      v16h bv = load_b16(Vp, Ss, t * 16, kc, lane);  // VT contiguous in keys
      acc[0][t] = wmma_f16(ap[0], bv, acc[0][t]);
      acc[1][t] = wmma_f16(ap[1], bv, acc[1][t]);
    }
  }

  // normalize + store merged-head layout [b][s][h*HD+d] (f16 for proj GEMM)
  int h = bh & (Hh - 1), bb = bh >> 4;
#pragma unroll
  for (int mi = 0; mi < 2; ++mi) {
#pragma unroll
    for (int t = 0; t < 4; ++t) {
      int n = h * HDd + t * 16 + lg;
#pragma unroll
      for (int i = 0; i < 8; ++i) {
        int r = q0 + mi * 16 + (hi << 3) + i;
        float v = acc[mi][t][i] / lrow[mi][i];
        O[((size_t)(bb * Ss + r)) * Dd + n] = (h16)v;
      }
    }
  }
}

// ---------------------------------------------------------------------------
// GEMM 2: out = attn(4096x1024) @ Wproj(1024x1024) + b   (fp32 output)
//   Same 32x64 register blocking as GEMM 1.
// ---------------------------------------------------------------------------
__global__ __launch_bounds__(256) void gemm_proj(
    const h16* __restrict__ A, const h16* __restrict__ Bt,
    const float* __restrict__ bias, float* __restrict__ out) {
  const int NTg = Dd / 64;                             // 16 column super-tiles
  int wave = threadIdx.x >> 5, lane = threadIdx.x & 31;
  int id = blockIdx.x * 8 + wave;
  int m0 = (id / NTg) * 32, n0 = (id % NTg) * 64;

  v8f acc[2][4];
#pragma unroll
  for (int mi = 0; mi < 2; ++mi)
#pragma unroll
    for (int j = 0; j < 4; ++j) acc[mi][j] = v8f{};

  for (int kk = 0; kk < Dd; kk += 32) {
    __builtin_prefetch(A + (size_t)(m0 + (lane & 15)) * Dd + kk + 64, 0, 0);
    __builtin_prefetch(Bt + (size_t)(n0 + (lane & 15)) * Dd + kk + 64, 0, 0);
    v16h a0 = load_a16(A, Dd, m0, kk, lane);
    v16h a1 = load_a16(A, Dd, m0 + 16, kk, lane);
#pragma unroll
    for (int j = 0; j < 4; ++j) {
      v16h b = load_b16(Bt, Dd, n0 + 16 * j, kk, lane);
      acc[0][j] = wmma_f16(a0, b, acc[0][j]);
      acc[1][j] = wmma_f16(a1, b, acc[1][j]);
    }
  }

  int lg = lane & 15, hi = lane >> 4;
#pragma unroll
  for (int j = 0; j < 4; ++j) {
    int n = n0 + 16 * j + lg;
    float bv = bias[n];
#pragma unroll
    for (int mi = 0; mi < 2; ++mi) {
#pragma unroll
      for (int i = 0; i < 8; ++i) {
        int r = m0 + mi * 16 + (hi << 3) + i;
        out[(size_t)r * Dd + n] = acc[mi][j][i] + bv;
      }
    }
  }
}

// ---------------------------------------------------------------------------
// launch
// ---------------------------------------------------------------------------
extern "C" void kernel_launch(void* const* d_in, const int* in_sizes, int n_in,
                              void* d_out, int out_size, void* d_ws,
                              size_t ws_size, hipStream_t stream) {
  const float* hidden   = (const float*)d_in[0];
  const float* c_attn_w = (const float*)d_in[1];
  const float* c_attn_b = (const float*)d_in[2];
  const float* c_proj_w = (const float*)d_in[3];
  const float* c_proj_b = (const float*)d_in[4];
  float* out = (float*)d_out;

  char* ws = (char*)d_ws;
  h16* hidden16 = (h16*)ws; ws += (size_t)MM * Dd * sizeof(h16);        // 8 MiB
  h16* wattnT   = (h16*)ws; ws += (size_t)3 * Dd * Dd * sizeof(h16);    // 6 MiB
  h16* q16      = (h16*)ws; ws += (size_t)MM * Dd * sizeof(h16);        // 8 MiB
  h16* k16      = (h16*)ws; ws += (size_t)MM * Dd * sizeof(h16);        // 8 MiB
  h16* vt16     = (h16*)ws; ws += (size_t)MM * Dd * sizeof(h16);        // 8 MiB
  h16* attn16   = (h16*)ws; ws += (size_t)MM * Dd * sizeof(h16);        // 8 MiB
  h16* wprojT   = (h16*)ws; ws += (size_t)Dd * Dd * sizeof(h16);        // 2 MiB

  // 1) precision staging
  {
    int n = MM * Dd;
    cvt_f16<<<(n + 255) / 256, 256, 0, stream>>>(hidden, hidden16, n);
  }
  cvt_transpose<<<(Dd * 3 * Dd + 255) / 256, 256, 0, stream>>>(
      c_attn_w, wattnT, Dd, 3 * Dd);
  cvt_transpose<<<(Dd * Dd + 255) / 256, 256, 0, stream>>>(
      c_proj_w, wprojT, Dd, Dd);

  // 2) QKV projection: (4096/32)*(3072/64) = 6144 wave-tiles / 8 = 768 blocks
  gemm_qkv<<<768, 256, 0, stream>>>(hidden16, wattnT, c_attn_b, q16, k16,
                                    vt16);

  // 3) flash attention: B*H*(S/32) = 2048 wave jobs / 8 = 256 blocks
  attn_flash<<<256, 256, 0, stream>>>(q16, k16, vt16, attn16);

  // 4) output projection: (4096/32)*(1024/64) = 2048 wave-tiles / 8 = 256 blk
  gemm_proj<<<256, 256, 0, stream>>>(attn16, wprojT, c_proj_b, out);
}
